// OrthonormalTimeSphericalNeRF_44203803411090
// MI455X (gfx1250) — matrix operand order repaired and checked by reading, not compiled, not run on hardware
//
#include <hip/hip_runtime.h>
#include <hip/hip_bf16.h>

// MI455X / gfx1250 fused kernel for OrthonormalTimeSphericalNeRF.
//
//   out[n,o] = BASE[o] + sum_{l,m,k,f} SH[n,lm] * Bes[n,l,k] * F[n,f] * W[o,lm,k,f]
//
// Factored as:  H[n, j=(o,lm,k)] = F[n,:17] @ Wf[:17, j]      -> v_wmma_f32_16x16x32_f16
//               out[n,o]        += G[n, lm*8+k] * H[n, j]     -> LDS + VALU
// with G = SH*Bes (360 per point) staged in LDS per 16-point wave tile.
//
// Column space is padded per output: 2 outputs x 368 columns = 23 tile pairs.
// The tile loop is kept ROLLED (unroll 1): one body processes the o=0 and o=1
// tiles of the same column range, sharing one set of G loads for both, and the
// B fragments land in fixed registers (no VGPR-MSB pressure / mov churn).

typedef __attribute__((ext_vector_type(16))) _Float16 v16h;
typedef __attribute__((ext_vector_type(8)))  float    v8f;

#define NPTS    32768
#define DFEAT   6120
#define TPO     23        // tiles per output (368 padded columns / 16)
#define NTILE   (2*TPO)   // 46 WMMA column tiles
#define GSTR    369       // padded G row stride (dwords): covers q<=367 and
                          // 8*369 % 64 == 8 -> no LDS bank conflict across halves

// ---------------------------------------------------------------------------
// Pack W [2 x 6120] f32  ->  B-operand fragments, f16.
// Wb[(tile*32 + lane)*16 + e] is the e-th half of `lane`'s v16h B register for
// column tile `tile`.  B is 32x16 (KxN): lane&15 = column n, lane>=16 holds
// K=8..15,24..31; e<8 -> K=8*(lane>>4)+e, e>=8 -> K=16+8*(lane>>4)+e-8.  Only K=f<=16 and q<360 are nonzero.
// ---------------------------------------------------------------------------
__global__ void pack_w_kernel(const float* __restrict__ W,
                              _Float16* __restrict__ Wb) {
    const int tile = blockIdx.x;          // 0..45
    const int lane = threadIdx.x;         // 0..31
    const int n    = lane & 15;
    const int kh   = lane >> 4;           // 0: K 0..7,16..23, 1: K 8..15,24..31
    const int o    = (tile >= TPO) ? 1 : 0;
    const int q    = (tile - TPO * o) * 16 + n;   // 0..367 (>=360 is padding)
    const int lm   = q >> 3;
    const int kb   = q & 7;
    const bool colok = (q < 360);
    const float* wrow = W + (size_t)o * DFEAT + (size_t)(lm * 8 + kb) * 17;
    _Float16* dst = Wb + (size_t)(tile * 32 + lane) * 16;
#pragma unroll
    for (int e = 0; e < 16; ++e) {
        int f = (e < 8) ? (kh * 8 + e) : (16 + kh * 8 + (e - 8)); // Fourier index (K)
        float v = (colok && f <= 16) ? wrow[f] : 0.0f;       // zero-pad K/q
        dst[e] = (_Float16)v;
    }
}

// ---------------------------------------------------------------------------
// Main fused kernel: 64 threads = 2 waves per block, 16 points per wave.
// ---------------------------------------------------------------------------
__launch_bounds__(64)
__global__ void nerf_kernel(const float* __restrict__ X,
                            const _Float16* __restrict__ Wb,
                            float* __restrict__ out) {
    __shared__ float Gs [2][16 * GSTR];   // G[p][q], q = lm*8+k, padded stride
    __shared__ float SHs[2][16 * 45];     // SH[p][l*9 + m+4]
    __shared__ float BEs[2][16 * 40];     // Bes[p][l*8 + k]
    __shared__ float ACC[2][32];          // out accumulator [p*2+o]

    const int wave = threadIdx.x >> 5;
    const int lane = threadIdx.x & 31;
    const int hi   = lane >> 4;           // 0 / 1 half-wave
    const int ln16 = lane & 15;
    const int base = blockIdx.x * 32 + wave * 16;   // first point of this wave

    float* G  = Gs [wave];
    float* SH = SHs[wave];
    float* BE = BEs[wave];
    float* AC = ACC[wave];

    AC[lane] = 0.0f;                      // zeroed before any barrier

    // ---- Fourier A fragment (16x32 f16, K=Fourier index, padded) ----------
    // ISA A layout: lanes 0-15 hold M=lane, e0..7 -> K 0..7, e8..15 -> K 16..23
    //               lanes 16-31 hold M=lane-16, e0..7 -> K 8..15, e8..15 -> K 24..31
    const float tt = X[(size_t)(base + ln16) * 4 + 3];
    const float wt = 6.28318530717958647692f * tt;
    float s[9], c[9];
    s[1] = __sinf(wt); c[1] = __cosf(wt);
#pragma unroll
    for (int n = 2; n <= 8; ++n) {
        s[n] = s[n - 1] * c[1] + c[n - 1] * s[1];
        c[n] = c[n - 1] * c[1] - s[n - 1] * s[1];
    }
    float av[16];
    if (hi == 0) {
        av[0] = 1.0f;                                  // f=0
#pragma unroll
        for (int n = 1; n <= 7; ++n) av[n] = s[n];     // f=1..7 (sin)
        av[8] = c[8];                                  // K=16 -> f=16 (cos n=8)
#pragma unroll
        for (int e = 9; e < 16; ++e) av[e] = 0.0f;     // K 17..23 pad
    } else {
        av[0] = s[8];                                  // K=8 -> f=8 (sin n=8)
#pragma unroll
        for (int n = 1; n <= 7; ++n) av[n] = c[n];     // f=9..15 (cos n=1..7)
#pragma unroll
        for (int e = 8; e < 16; ++e) av[e] = 0.0f;     // K 24..31 pad
    }
    v16h a;
#pragma unroll
    for (int e = 0; e < 16; ++e) a[e] = (_Float16)av[e];

    // ---- per-point bases: lanes 0-15 do SH, lanes 16-31 do Bessel ---------
    {
        const int p = ln16;
        const float x0 = X[(size_t)(base + p) * 4 + 0];
        const float y0 = X[(size_t)(base + p) * 4 + 1];
        const float z0 = X[(size_t)(base + p) * 4 + 2];
        const float r2xy = x0 * x0 + y0 * y0;
        const float r    = sqrtf(r2xy + z0 * z0);
        if (hi == 0) {
            // --- real spherical harmonics up to L=4 ---
            const float rxy  = sqrtf(r2xy);
            const float invr = 1.0f / r;
            const float ct = z0 * invr;
            const float st = rxy * invr;
            const float irxy = (rxy > 1e-30f) ? (1.0f / rxy) : 0.0f;
            const float cp = (rxy > 1e-30f) ? (x0 * irxy) : 1.0f;
            const float sp = y0 * irxy;
            float P[5][5];
            P[0][0] = 1.0f;
#pragma unroll
            for (int m = 1; m <= 4; ++m) P[m][m] = P[m - 1][m - 1] * st * (float)(2 * m - 1);
#pragma unroll
            for (int m = 0; m <= 3; ++m) P[m + 1][m] = (float)(2 * m + 1) * ct * P[m][m];
#pragma unroll
            for (int m = 0; m <= 2; ++m)
#pragma unroll
                for (int l = m + 2; l <= 4; ++l)
                    P[l][m] = ((float)(2 * l - 1) * ct * P[l - 1][m]
                             - (float)(l + m - 1) * P[l - 2][m]) * (1.0f / (float)(l - m));
            float cm[5], sm[5];
            cm[0] = 1.0f; sm[0] = 0.0f; cm[1] = cp; sm[1] = sp;
#pragma unroll
            for (int m = 2; m <= 4; ++m) {
                cm[m] = cm[m - 1] * cp - sm[m - 1] * sp;
                sm[m] = sm[m - 1] * cp + cm[m - 1] * sp;
            }
#pragma unroll
            for (int l = 0; l <= 4; ++l) {
#pragma unroll
                for (int mm = 0; mm < 9; ++mm) SH[p * 45 + l * 9 + mm] = 0.0f;
#pragma unroll
                for (int m = 0; m <= l; ++m) {
                    float ratio = 1.0f;
#pragma unroll
                    for (int i = l - m + 1; i <= l + m; ++i) ratio *= (float)i;
                    const float norm = sqrtf((float)(2 * l + 1) * 0.07957747154594767f / ratio);
                    if (m == 0) {
                        SH[p * 45 + l * 9 + 4] = norm * P[l][0];
                    } else {
                        const float nv = 1.4142135623730951f * norm * P[l][m];
                        SH[p * 45 + l * 9 + 4 + m] = nv * cm[m];
                        SH[p * 45 + l * 9 + 4 - m] = nv * sm[m];
                    }
                }
            }
        } else {
            // --- spherical Bessel j_0..j_4 via Miller downward recurrence ---
#pragma unroll
            for (int kb = 0; kb < 8; ++kb) {
                const float kk  = (float)(kb + 1);
                const float rk  = r * kk;
                const float inv = 1.0f / rk;
                float j2 = 0.0f, j1 = 1.0f, yv[5];
#pragma unroll
                for (int i = 12; i >= 1; --i) {
                    const float j0 = (float)(2 * i + 1) * inv * j1 - j2;
                    if (i - 1 < 5) yv[i - 1] = j0;
                    j2 = j1; j1 = j0;
                }
                const float tj0 = __sinf(rk) * inv;                 // true j0
                const float scl = tj0 / yv[0] * 0.7978845608028654f * kk; // *sqrt(2/pi)*k
#pragma unroll
                for (int l = 0; l < 5; ++l) BE[p * 40 + l * 8 + kb] = yv[l] * scl;
            }
        }
    }
    __syncthreads();

    // ---- G[p][q] = SH[p][lm] * Bes[p][l*8+k];  q in [360,368) zeroed -------
    // q-block outer so the (lm, l, kb) decode is done once per 32 columns;
    // inner p loop unrolled -> LDS offsets become immediates.
#pragma unroll 1
    for (int q0 = 0; q0 < 368; q0 += 32) {
        const int q = q0 + lane;
        if (q < 360) {
            const int lm = q >> 3;
            const int l  = (int)((unsigned)lm / 9u);
            const int sb = lm;                 // SH col
            const int bb = l * 8 + (q & 7);    // BE col
#pragma unroll
            for (int p = 0; p < 16; ++p)
                G[p * GSTR + q] = SH[p * 45 + sb] * BE[p * 40 + bb];
        } else if (q < 368) {
#pragma unroll
            for (int p = 0; p < 16; ++p)
                G[p * GSTR + q] = 0.0f;        // pad columns must be finite
        }
    }
    __syncthreads();

    // ---- 23 tile pairs; one set of G loads feeds both output accumulators --
    float acc0[8], acc1[8];
#pragma unroll
    for (int rr = 0; rr < 8; ++rr) { acc0[rr] = 0.0f; acc1[rr] = 0.0f; }

    const v16h* bp = (const v16h*)Wb + lane;   // +32 v16h per tile
    const float* gbase = &G[hi * 8 * GSTR];
    const v8f cz = {0.f, 0.f, 0.f, 0.f, 0.f, 0.f, 0.f, 0.f};

#pragma unroll 1
    for (int t = 0; t < TPO; ++t) {
        const v16h b0 = bp[t * 32];                 // o = 0 tile
        const v16h b1 = bp[(TPO + t) * 32];         // o = 1 tile, same columns
        v8f d0 = __builtin_amdgcn_wmma_f32_16x16x32_f16(
            false, a, false, b0, (short)0, cz, false, false);
        v8f d1 = __builtin_amdgcn_wmma_f32_16x16x32_f16(
            false, a, false, b1, (short)0, cz, false, false);
        const float* g = gbase + (t * 16 + ln16);   // q = t*16 + ln16 (<368)
#pragma unroll
        for (int rr = 0; rr < 8; ++rr) {            // row M = rr + 8*hi
            const float gv = g[rr * GSTR];          // shared by both outputs
            acc0[rr] += d0[rr] * gv;
            acc1[rr] += d1[rr] * gv;
        }
    }

    // ---- reduce across the 16 column-lanes per point via LDS atomics -------
#pragma unroll
    for (int rr = 0; rr < 8; ++rr) {
        const int p = hi * 8 + rr;
        atomicAdd(&AC[p * 2 + 0], acc0[rr]);
        atomicAdd(&AC[p * 2 + 1], acc1[rr]);
    }
    __syncthreads();

    // lane enumerates p*2+o for this wave's 16 points
    const float res = AC[lane] + ((lane & 1) ? 5.0f : 10.0f);  // +BASE_LOG_{T,D}
    out[(size_t)base * 2 + lane] = res;
}

// ---------------------------------------------------------------------------
extern "C" void kernel_launch(void* const* d_in, const int* in_sizes, int n_in,
                              void* d_out, int out_size, void* d_ws, size_t ws_size,
                              hipStream_t stream) {
    const float* X = (const float*)d_in[0];     // [32768, 4] f32
    const float* W = (const float*)d_in[1];     // [2, 6120] f32
    float* out = (float*)d_out;                 // [32768, 2] f32
    _Float16* Wb = (_Float16*)d_ws;             // 46*32*16 f16 = 47104 B

    pack_w_kernel<<<NTILE, 32, 0, stream>>>(W, Wb);
    nerf_kernel<<<NPTS / 32, 64, 0, stream>>>(X, Wb, out);
}
